// SlotAttention_44719199486530
// MI455X (gfx1250) — matrix operand order repaired and checked by reading, not compiled, not run on hardware
//
#include <hip/hip_runtime.h>
#include <hip/hip_bf16.h>

// ---------------------------------------------------------------------------
// Slot Attention for MI455X (gfx1250): f16 WMMA GEMMs, f32 accumulation.
// Operand layouts chosen so every WMMA fragment = two 128-bit loads.
// B=64, N=4096, D_IN=128, D_SLOTS=64, K=11, T=3, D_HID=256
// ---------------------------------------------------------------------------

#define BB    64
#define NN    4096
#define DIN   128
#define DS    64
#define KS    11
#define KPAD  16
#define DHID  256
#define TITER 3
#define EPS   1e-8f

typedef __attribute__((ext_vector_type(16))) _Float16 v16h;
typedef __attribute__((ext_vector_type(8)))  _Float16 v8h;
typedef __attribute__((ext_vector_type(8)))  float    v8f;
typedef __attribute__((ext_vector_type(4)))  int      v4i;

#define AS_GLOBAL __attribute__((address_space(1)))
#define AS_LDS    __attribute__((address_space(3)))

#if defined(__has_builtin)
#if __has_builtin(__builtin_amdgcn_global_load_async_to_lds_b128) && \
    __has_builtin(__builtin_amdgcn_s_wait_asynccnt)
#define USE_ASYNC_LDS 1
#endif
#endif

__device__ __forceinline__ v16h pack16(v8h lo, v8h hi) {
  return __builtin_shufflevector(lo, hi, 0, 1, 2, 3, 4, 5, 6, 7,
                                 8, 9, 10, 11, 12, 13, 14, 15);
}

// A (16x32 f16) fragment from a ROW-MAJOR tile (ld = halves per row).
// lane L: M=L&15, hi=L>>4; halves 0..7 = tile[M][8hi..8hi+7],
//                          halves 8..15 = tile[M][16+8hi..16+8hi+7]
__device__ __forceinline__ v16h frag_A_rm(const _Float16* tile, int ld) {
  int lane = threadIdx.x & 31;
  int M = lane & 15, hi = lane >> 4;
  const _Float16* p = tile + M * ld + 8 * hi;
  return pack16(*(const v8h*)(p), *(const v8h*)(p + 16));
}

// B (32x16 f16) fragment from a TRANSPOSED tile: tileT[col][K] (ld halves).
// lane L: N=L&15, hi=L>>4; halves 0..7 = tileT[N][8hi..8hi+7], 8..15 = tileT[N][16+8hi..16+8hi+7]
__device__ __forceinline__ v16h frag_B_cm(const _Float16* tileT, int ld) {
  int lane = threadIdx.x & 31;
  int N = lane & 15, hi = lane >> 4;
  const _Float16* p = tileT + N * ld + 8 * hi;
  return pack16(*(const v8h*)(p), *(const v8h*)(p + 16));
}

#define WMMA_F16(a, b, c) \
  __builtin_amdgcn_wmma_f32_16x16x32_f16(false, (a), false, (b), (short)0, (c), false, false)

// ---------------------------------------------------------------------------
// K0: broadcast init_latents -> ws slots [B,K,DS]
// ---------------------------------------------------------------------------
__global__ void k_init_slots(const float* __restrict__ init_latents,
                             float* __restrict__ slots) {
  int i = blockIdx.x * blockDim.x + threadIdx.x;
  if (i < BB * KS * DS) slots[i] = init_latents[i % (KS * DS)];
}

// ---------------------------------------------------------------------------
// K1: fused LayerNorm(inputs) + keys = xn@Wk (row-major out),
//     vals = xn@Wv (TRANSPOSED out: [B][c][n]).
// block = 256 (8 waves); one 16-row tile per wave. Weights staged transposed.
// ---------------------------------------------------------------------------
__global__ __launch_bounds__(256) void k_ln_kv(
    const float* __restrict__ x,                 // [B*N, 128]
    const float* __restrict__ g, const float* __restrict__ bet,
    const float* __restrict__ Wk, const float* __restrict__ Wv,  // [128,64]
    _Float16* __restrict__ keysH,                // [B*N,64] row-major
    _Float16* __restrict__ valsT)                // [B,64,4096] transposed
{
  __shared__ __align__(16) _Float16 wKt[DS * DIN];   // [c][k]  16 KB
  __shared__ __align__(16) _Float16 wVt[DS * DIN];   // [c][k]  16 KB
  __shared__ __align__(16) _Float16 xn[8][16 * DIN]; // row-major per-wave, 32 KB

  int tid = threadIdx.x;
  for (int i = tid; i < DIN * DS; i += 256) {
    int k = i >> 6, c = i & 63;            // Wk row-major [k][c]
    wKt[c * DIN + k] = (_Float16)Wk[i];
    wVt[c * DIN + k] = (_Float16)Wv[i];
  }
  __syncthreads();

  int wave = tid >> 5, lane = tid & 31;
  size_t row0 = ((size_t)blockIdx.x * 8 + wave) * 16;

  // CDNA5 prefetch path for the next tile of inputs
  __builtin_prefetch(x + (row0 + 128) * DIN, 0, 3);

  // LayerNorm 16 rows (4 elements/lane, wave32 shuffle reductions)
  for (int r = 0; r < 16; ++r) {
    const float* xr = x + (row0 + r) * DIN;
    float v0 = xr[lane], v1 = xr[lane + 32], v2 = xr[lane + 64], v3 = xr[lane + 96];
    float s = v0 + v1 + v2 + v3;
#pragma unroll
    for (int m = 16; m >= 1; m >>= 1) s += __shfl_xor(s, m);
    float mean = s * (1.0f / 128.0f);
    float d0 = v0 - mean, d1 = v1 - mean, d2 = v2 - mean, d3 = v3 - mean;
    float q = d0 * d0 + d1 * d1 + d2 * d2 + d3 * d3;
#pragma unroll
    for (int m = 16; m >= 1; m >>= 1) q += __shfl_xor(q, m);
    float rstd = rsqrtf(q * (1.0f / 128.0f) + 1e-5f);
    _Float16* dst = &xn[wave][r * DIN];
    dst[lane]      = (_Float16)(d0 * rstd * g[lane]      + bet[lane]);
    dst[lane + 32] = (_Float16)(d1 * rstd * g[lane + 32] + bet[lane + 32]);
    dst[lane + 64] = (_Float16)(d2 * rstd * g[lane + 64] + bet[lane + 64]);
    dst[lane + 96] = (_Float16)(d3 * rstd * g[lane + 96] + bet[lane + 96]);
  }

  // GEMM: 4 K-chunks of 32 x 4 col tiles, keys + vals
  v8f accK[4] = {};
  v8f accV[4] = {};
#pragma unroll
  for (int kc = 0; kc < 4; ++kc) {
    v16h a = frag_A_rm(&xn[wave][kc * 32], DIN);
#pragma unroll
    for (int ct = 0; ct < 4; ++ct) {
      v16h bk = frag_B_cm(&wKt[(ct * 16) * DIN + kc * 32], DIN);
      accK[ct] = WMMA_F16(a, bk, accK[ct]);
      v16h bv = frag_B_cm(&wVt[(ct * 16) * DIN + kc * 32], DIN);
      accV[ct] = WMMA_F16(a, bv, accV[ct]);
    }
  }

  int N = lane & 15, hi = lane >> 4;
  int b = (int)(row0 >> 12);          // tiles never straddle a batch (4096%16==0)
  int nloc = (int)(row0 & 4095);
#pragma unroll
  for (int ct = 0; ct < 4; ++ct) {
#pragma unroll
    for (int r = 0; r < 8; ++r) {
      int M = r + (hi ? 8 : 0);
      keysH[(row0 + M) * DS + ct * 16 + N] = (_Float16)accK[ct][r];
      valsT[((size_t)b * DS + ct * 16 + N) * NN + nloc + M] = (_Float16)accV[ct][r];
    }
  }
}

// ---------------------------------------------------------------------------
// K2: q = LN(slots) @ Wq (one wave per batch); zero colsum.
// qH: [B,16,64] f16 row-major ([slot][channel] == B-operand transposed form).
// ---------------------------------------------------------------------------
__global__ __launch_bounds__(32) void k_q(
    const float* __restrict__ slots,   // [B,11,64]
    const float* __restrict__ gq, const float* __restrict__ bq,
    const float* __restrict__ Wq,      // [64,64]
    _Float16* __restrict__ qH,         // [B,16,64]
    float* __restrict__ colsum)        // [B,16]
{
  int b = blockIdx.x, lane = threadIdx.x;
  __shared__ float lnrow[KS][DS];
  for (int k = 0; k < KS; ++k) {
    const float* sr = slots + ((size_t)b * KS + k) * DS;
    float v0 = sr[lane], v1 = sr[lane + 32];
    float s = v0 + v1;
#pragma unroll
    for (int m = 16; m >= 1; m >>= 1) s += __shfl_xor(s, m);
    float mean = s * (1.0f / 64.0f);
    float d0 = v0 - mean, d1 = v1 - mean;
    float q2 = d0 * d0 + d1 * d1;
#pragma unroll
    for (int m = 16; m >= 1; m >>= 1) q2 += __shfl_xor(q2, m);
    float rstd = rsqrtf(q2 * (1.0f / 64.0f) + 1e-5f);
    lnrow[k][lane]      = d0 * rstd * gq[lane]      + bq[lane];
    lnrow[k][lane + 32] = d1 * rstd * gq[lane + 32] + bq[lane + 32];
  }
  for (int k = 0; k < KPAD; ++k) {
    for (int c = lane; c < DS; c += 32) {
      float acc = 0.0f;
      if (k < KS) {
        for (int d = 0; d < DS; ++d) acc += lnrow[k][d] * Wq[d * DS + c];
      }
      qH[((size_t)b * KPAD + k) * DS + c] = (_Float16)acc;
    }
  }
  if (lane < KPAD) colsum[b * KPAD + lane] = 0.0f;
}

// ---------------------------------------------------------------------------
// K3: logits = scale * keys @ q^T (WMMA), masked softmax over 11 slots, +EPS;
//     writes attn -> d_out, attn^T f16 -> ws, atomic colsum accumulation.
// grid = (N/128, B), block = 256 (8 waves x 16 rows).
// Key tiles staged to LDS via async global->LDS when the toolchain has it.
// ---------------------------------------------------------------------------
__global__ __launch_bounds__(256) void k_attn(
    const _Float16* __restrict__ keysH, // [B*N,64]
    const _Float16* __restrict__ qH,    // [B,16,64]
    float* __restrict__ attn_out,       // [B,N,11]
    _Float16* __restrict__ attnT,       // [B,16,4096]
    float* __restrict__ colsum)         // [B,16]
{
  int b = blockIdx.y;
  __shared__ __align__(16) _Float16 qs[KPAD * DS];    // [slot][channel], 2 KB
  __shared__ __align__(16) _Float16 kst[8][16 * DS];  // per-wave key tiles, 16 KB

  int tid = threadIdx.x;
  {
    const v8h* src = (const v8h*)(qH + (size_t)b * KPAD * DS);
    v8h* dst = (v8h*)qs;
    for (int i = tid; i < (KPAD * DS) / 8; i += 256) dst[i] = src[i];
  }
  __syncthreads();

  int wave = tid >> 5, lane = tid & 31;
  int n0 = (blockIdx.x * 8 + wave) * 16;
  size_t rowbase = (size_t)b * NN + n0;

  // stage 16x64 f16 key tile (rows contiguous -> 2048 contiguous bytes)
  _Float16* kt = &kst[wave][0];
  const _Float16* src = keysH + rowbase * DS;
#if defined(USE_ASYNC_LDS)
#pragma unroll
  for (int i = 0; i < 4; ++i) {
    __builtin_amdgcn_global_load_async_to_lds_b128(
        (AS_GLOBAL v4i*)(src + (i * 32 + lane) * 8),
        (AS_LDS v4i*)(kt + (i * 32 + lane) * 8),
        0, 0);
  }
  __builtin_amdgcn_s_wait_asynccnt(0);
#else
  {
    const float4* s4 = (const float4*)src;
    float4* d4 = (float4*)kt;
    for (int i = lane; i < 128; i += 32) d4[i] = s4[i];
  }
#endif

  v8f c0 = {};
  v16h a0 = frag_A_rm(kt, DS);
  v16h a1 = frag_A_rm(kt + 32, DS);
  v16h b0 = frag_B_cm(qs, DS);            // chunk c=0..31
  v16h b1 = frag_B_cm(qs + 32, DS);       // chunk c=32..63
  c0 = WMMA_F16(a0, b0, c0);
  c0 = WMMA_F16(a1, b1, c0);

  const float scale = 0.125f;  // 64^-0.5
  int N = lane & 15, hi = lane >> 4;
  float colpart = 0.0f;
#pragma unroll
  for (int r = 0; r < 8; ++r) {
    float L = (N < KS) ? c0[r] * scale : -3.0e38f;
    float mx = L;
#pragma unroll
    for (int s = 1; s < 16; s <<= 1) mx = fmaxf(mx, __shfl_xor(mx, s));
    float ex = (N < KS) ? __expf(L - mx) : 0.0f;
    float ssum = ex;
#pragma unroll
    for (int s = 1; s < 16; s <<= 1) ssum += __shfl_xor(ssum, s);
    float attn = ex / ssum + EPS;
    int M = r + (hi ? 8 : 0);
    if (N < KS) {
      attn_out[((size_t)b * NN + n0 + M) * KS + N] = attn;
      colpart += attn;
    }
    attnT[((size_t)b * KPAD + N) * NN + n0 + M] = (_Float16)((N < KS) ? attn : 0.0f);
  }
  if (N < KS) atomicAdd(&colsum[b * KPAD + N], colpart);
}

// ---------------------------------------------------------------------------
// K4: updates = (attn^T @ vals) / colsum.
// grid = B, block = 256 (8 waves split the N range), LDS cross-wave reduce.
// A from attnT (row-major over n), B from valsT ([c][n]) -> all b128 loads.
// ---------------------------------------------------------------------------
__global__ __launch_bounds__(256) void k_updates(
    const _Float16* __restrict__ attnT,  // [B,16,4096]
    const _Float16* __restrict__ valsT,  // [B,64,4096]
    const float* __restrict__ colsum,    // [B,16]
    float* __restrict__ updates)         // [B,16,64]
{
  int b = blockIdx.x, tid = threadIdx.x;
  int wave = tid >> 5, lane = tid & 31;
  int N = lane & 15, hi = lane >> 4;
  __shared__ float red[8][32][32];       // 32 KB

  v8f acc[4] = {};
  const _Float16* at = attnT + (size_t)b * KPAD * NN;
  const _Float16* vb = valsT + (size_t)b * DS * NN;

  int nbeg = wave * (NN / 8), nend = nbeg + (NN / 8);
  for (int n0 = nbeg; n0 < nend; n0 += 32) {
    const _Float16* pa = at + (size_t)N * NN + n0 + 8 * hi;
    v16h a = pack16(*(const v8h*)pa, *(const v8h*)(pa + 16));
#pragma unroll
    for (int ct = 0; ct < 4; ++ct) {
      const _Float16* pb = vb + (size_t)(ct * 16 + N) * NN + n0 + 8 * hi;
      v16h bf = pack16(*(const v8h*)pb, *(const v8h*)(pb + 16));
      acc[ct] = WMMA_F16(a, bf, acc[ct]);
    }
  }

#pragma unroll
  for (int ct = 0; ct < 4; ++ct)
#pragma unroll
    for (int r = 0; r < 8; ++r) red[wave][lane][ct * 8 + r] = acc[ct][r];
  __syncthreads();

  if (wave == 0) {
#pragma unroll
    for (int ct = 0; ct < 4; ++ct) {
#pragma unroll
      for (int r = 0; r < 8; ++r) {
        float s = 0.0f;
#pragma unroll
        for (int w = 0; w < 8; ++w) s += red[w][lane][ct * 8 + r];
        int M = r + (hi ? 8 : 0);
        float u = (M < KS) ? s / colsum[b * KPAD + M] : 0.0f;
        updates[((size_t)b * KPAD + M) * DS + ct * 16 + N] = u;
      }
    }
  }
}

// ---------------------------------------------------------------------------
// K5: GRU cell + LN + MLP residual. One wave per (b,k) row of 64.
// ---------------------------------------------------------------------------
__global__ __launch_bounds__(32) void k_gru_mlp(
    const float* __restrict__ updates,                 // [B,16,64]
    float* __restrict__ slots,                         // [B,11,64] (in/out)
    const float* __restrict__ w_ih, const float* __restrict__ w_hh,  // [64,192]
    const float* __restrict__ b_ih, const float* __restrict__ b_hh,  // [192]
    const float* __restrict__ lng, const float* __restrict__ lnb,    // [64]
    const float* __restrict__ W1, const float* __restrict__ b1,      // [64,256],[256]
    const float* __restrict__ W2, const float* __restrict__ b2,      // [256,64],[64]
    float* __restrict__ out_slots, int write_out)
{
  int idx = blockIdx.x;
  int b = idx / KS, k = idx % KS;
  int lane = threadIdx.x;
  __shared__ float xrow[DS], hrow[DS], gx[3 * DS], gh[3 * DS], srow[DS],
                   lnf[DS], h1[DHID];

  const float* up = updates + ((size_t)b * KPAD + k) * DS;
  float* sp = slots + ((size_t)b * KS + k) * DS;
  xrow[lane] = up[lane];       xrow[lane + 32] = up[lane + 32];
  hrow[lane] = sp[lane];       hrow[lane + 32] = sp[lane + 32];
  __syncthreads();  // single-wave workgroup: S_NOP; LDS in-order

  for (int j = lane; j < 3 * DS; j += 32) {
    float ax = b_ih[j], ah = b_hh[j];
    for (int c = 0; c < DS; ++c) {
      ax += xrow[c] * w_ih[c * 3 * DS + j];
      ah += hrow[c] * w_hh[c * 3 * DS + j];
    }
    gx[j] = ax; gh[j] = ah;
  }

#pragma unroll
  for (int t = 0; t < 2; ++t) {
    int c = lane + t * 32;
    float r = 1.0f / (1.0f + __expf(-(gx[c] + gh[c])));
    float z = 1.0f / (1.0f + __expf(-(gx[DS + c] + gh[DS + c])));
    float nn = tanhf(gx[2 * DS + c] + r * gh[2 * DS + c]);
    srow[c] = (1.0f - z) * nn + z * hrow[c];
  }

  float v0 = srow[lane], v1 = srow[lane + 32];
  float s = v0 + v1;
#pragma unroll
  for (int m = 16; m >= 1; m >>= 1) s += __shfl_xor(s, m);
  float mean = s * (1.0f / 64.0f);
  float d0 = v0 - mean, d1 = v1 - mean;
  float q2 = d0 * d0 + d1 * d1;
#pragma unroll
  for (int m = 16; m >= 1; m >>= 1) q2 += __shfl_xor(q2, m);
  float rstd = rsqrtf(q2 * (1.0f / 64.0f) + 1e-5f);
  lnf[lane]      = d0 * rstd * lng[lane]      + lnb[lane];
  lnf[lane + 32] = d1 * rstd * lng[lane + 32] + lnb[lane + 32];

  for (int j = lane; j < DHID; j += 32) {
    float a = b1[j];
    for (int c = 0; c < DS; ++c) a += lnf[c] * W1[c * DHID + j];
    h1[j] = fmaxf(a, 0.0f);
  }

#pragma unroll
  for (int t = 0; t < 2; ++t) {
    int c = lane + t * 32;
    float a = b2[c];
    for (int j = 0; j < DHID; ++j) a += h1[j] * W2[j * DS + c];
    float outv = srow[c] + a;
    sp[c] = outv;
    if (write_out) out_slots[((size_t)b * KS + k) * DS + c] = outv;
  }
}

// ---------------------------------------------------------------------------
// Host launcher
// ---------------------------------------------------------------------------
extern "C" void kernel_launch(void* const* d_in, const int* in_sizes, int n_in,
                              void* d_out, int out_size, void* d_ws, size_t ws_size,
                              hipStream_t stream) {
  const float* inputs       = (const float*)d_in[0];
  const float* ln_in_g      = (const float*)d_in[1];
  const float* ln_in_b      = (const float*)d_in[2];
  const float* Wk           = (const float*)d_in[3];
  const float* Wv           = (const float*)d_in[4];
  const float* init_latents = (const float*)d_in[5];
  const float* ln_q_g       = (const float*)d_in[6];
  const float* ln_q_b       = (const float*)d_in[7];
  const float* Wq           = (const float*)d_in[8];
  const float* w_ih         = (const float*)d_in[9];
  const float* w_hh         = (const float*)d_in[10];
  const float* b_ih         = (const float*)d_in[11];
  const float* b_hh         = (const float*)d_in[12];
  const float* ln_f_g       = (const float*)d_in[13];
  const float* ln_f_b       = (const float*)d_in[14];
  const float* W1           = (const float*)d_in[15];
  const float* b1           = (const float*)d_in[16];
  const float* W2           = (const float*)d_in[17];
  const float* b2           = (const float*)d_in[18];

  float* out_slots = (float*)d_out;                        // [B,K,64]
  float* out_attn  = (float*)d_out + (size_t)BB * KS * DS; // [B,N,K]

  // workspace carve-up (bytes)
  char* ws = (char*)d_ws;
  _Float16* keysH  = (_Float16*)(ws);                       // 32 MB
  _Float16* valsT  = (_Float16*)(ws + 33554432ull);         // 32 MB
  _Float16* attnT  = (_Float16*)(ws + 67108864ull);         // 8 MB
  _Float16* qH     = (_Float16*)(ws + 75497472ull);         // 128 KB
  float*    colsum = (float*)   (ws + 75628544ull);         // 4 KB
  float*    upds   = (float*)   (ws + 75632640ull);         // 256 KB
  float*    slots  = (float*)   (ws + 75894784ull);         // 176 KB

  k_init_slots<<<(BB * KS * DS + 255) / 256, 256, 0, stream>>>(init_latents, slots);

  // 262144 rows / 128 rows per block
  k_ln_kv<<<2048, 256, 0, stream>>>(inputs, ln_in_g, ln_in_b, Wk, Wv, keysH, valsT);

  for (int t = 0; t < TITER; ++t) {
    k_q<<<BB, 32, 0, stream>>>(slots, ln_q_g, ln_q_b, Wq, qH, colsum);
    k_attn<<<dim3(NN / 128, BB), 256, 0, stream>>>(keysH, qH, out_attn, attnT, colsum);
    k_updates<<<BB, 256, 0, stream>>>(attnT, valsT, colsum, upds);
    k_gru_mlp<<<BB * KS, 32, 0, stream>>>(upds, slots, w_ih, w_hh, b_ih, b_hh,
                                          ln_f_g, ln_f_b, W1, b1, W2, b2,
                                          out_slots, (t == TITER - 1) ? 1 : 0);
  }
}
